// Attention_10161892622384
// MI455X (gfx1250) — matrix operand order repaired and checked by reading, not compiled, not run on hardware
//
#include <hip/hip_runtime.h>
#include <hip/hip_bf16.h>
#include <math.h>

// ---------------------------------------------------------------------------
// CDNA5 (gfx1250) wave32 WMMA attention pipeline, bf16 compute / f32 accum.
// GEMMs use async global->LDS DMA (ASYNCcnt) with double-buffered tiles.
// ---------------------------------------------------------------------------

typedef __attribute__((ext_vector_type(16))) __bf16 v16bf;
typedef __attribute__((ext_vector_type(8)))  __bf16 v8bf;
typedef __attribute__((ext_vector_type(8)))  float  v8f;

union BFrag { v16bf v; v8bf h[2]; };

__device__ __forceinline__ unsigned short f2bf(float f) {
  union { float f; unsigned u; } cv; cv.f = f;
  unsigned u = cv.u;
  unsigned r = u + 0x7FFFu + ((u >> 16) & 1u);   // round-to-nearest-even
  return (unsigned short)(r >> 16);
}

__device__ __forceinline__ v8f wmma_bf16(v16bf a, v16bf b, v8f c) {
  // D = A(16x32 bf16) * B(32x16 bf16) + C(16x16 f32)
  return __builtin_amdgcn_wmma_f32_16x16x32_bf16(
      false, a, false, b, (short)0, c, false, false);
}

// Async global->LDS DMA, 16 bytes/lane, GVS mode (SGPR64 base + VGPR32 offset).
// Tracked by ASYNCcnt; no VGPR data round-trip.
#define ASYNC_B128(ldsaddr, base64, off32)                          \
  asm volatile("global_load_async_to_lds_b128 %0, %1, %2"           \
               :: "v"(ldsaddr), "v"(off32), "s"(base64) : "memory")

#define WAIT_ASYNC0() asm volatile("s_wait_asynccnt 0x0" ::: "memory")

// XOR-lane swizzle within 32 lanes (and_mask=0x1F, or=0, xor=m)
#define SWZ_F(x, m) \
  __int_as_float(__builtin_amdgcn_ds_swizzle(__float_as_int(x), (((m) << 10) | 0x1F)))

__device__ __forceinline__ float red16_max(float x) {
  x = fmaxf(x, SWZ_F(x, 1)); x = fmaxf(x, SWZ_F(x, 2));
  x = fmaxf(x, SWZ_F(x, 4)); x = fmaxf(x, SWZ_F(x, 8));
  return x;
}
__device__ __forceinline__ float red16_sum(float x) {
  x += SWZ_F(x, 1); x += SWZ_F(x, 2);
  x += SWZ_F(x, 4); x += SWZ_F(x, 8);
  return x;
}

// ---------------------------------------------------------------------------
// Stage 0: f32 -> bf16 conversions
// ---------------------------------------------------------------------------
__global__ __launch_bounds__(256) void k_cvt(const float* __restrict__ in,
                                             unsigned short* __restrict__ out, int n) {
  int i = blockIdx.x * 256 + threadIdx.x;
  if (i < n) out[i] = f2bf(in[i]);
}

// W[k][n] (KxN) -> Wt[n][k] (NxK) bf16, so B-fragments are k-contiguous loads
__global__ __launch_bounds__(256) void k_cvt_wt(const float* __restrict__ W,
                                                unsigned short* __restrict__ Wt,
                                                int K, int N) {
  int i = blockIdx.x * 256 + threadIdx.x;
  if (i < K * N) {
    int k = i / N, n = i % N;
    Wt[(size_t)n * K + k] = f2bf(W[i]);
  }
}

// ---------------------------------------------------------------------------
// Generic WMMA GEMM:  C[M x N] = A[M x K](bf16,row-major) * Wt[N x K]^T + bias
//   mode 0: store bf16 row-major [M x N]
//   mode 1: store f32  row-major [M x N]
//   mode 2: store bf16 per-head transposed: out[((b*H+h)*64+dim)*S + s]
// Block tile 128x128, BK=32, 256 threads = 8 waves (2M x 4N), wave = 64x32.
// Double-buffered LDS tiles filled by async global->LDS DMA.
// ---------------------------------------------------------------------------
__global__ __launch_bounds__(256) void k_gemm_bf16(
    const unsigned short* __restrict__ A, const unsigned short* __restrict__ Bt,
    const float* __restrict__ bias, void* __restrict__ Cout,
    int M, int N, int K, int mode, int H, int S) {
  __shared__ __attribute__((aligned(32))) unsigned short lds_a[2][128 * 32];
  __shared__ __attribute__((aligned(32))) unsigned short lds_b[2][128 * 32];

  const int wave = threadIdx.x >> 5, lane = threadIdx.x & 31;
  const int wm = wave >> 2, wn = wave & 3;
  const int m0 = blockIdx.x * 128, n0 = blockIdx.y * 128;
  const int mrow = lane & 15, khalf = lane >> 4;

  v8f acc[4][2];
#pragma unroll
  for (int i = 0; i < 4; i++)
#pragma unroll
    for (int j = 0; j < 2; j++) acc[i][j] = (v8f){0, 0, 0, 0, 0, 0, 0, 0};

  // Each thread DMAs one 32B chunk of A and of Bt per tile (2x b128 each).
  const int ldr = threadIdx.x >> 1, seg = threadIdx.x & 1;
  const unsigned la = (unsigned)(uintptr_t)&lds_a[0][ldr * 32 + seg * 16];
  const unsigned lb = (unsigned)(uintptr_t)&lds_b[0][ldr * 32 + seg * 16];
  const unsigned long long abase = (unsigned long long)(uintptr_t)A;
  const unsigned long long bbase = (unsigned long long)(uintptr_t)Bt;
  const unsigned aoff = (unsigned)(((size_t)(m0 + ldr) * K + seg * 16) * 2);
  const unsigned boff = (unsigned)(((size_t)(n0 + ldr) * K + seg * 16) * 2);

  auto stage = [&](int buf, int k0) {
    const unsigned lab = la + (unsigned)buf * 8192u;
    const unsigned lbb = lb + (unsigned)buf * 8192u;
    const unsigned ao = aoff + (unsigned)k0 * 2u;
    const unsigned bo = boff + (unsigned)k0 * 2u;
    ASYNC_B128(lab,       abase, ao);
    ASYNC_B128(lab + 16u, abase, ao + 16u);
    ASYNC_B128(lbb,       bbase, bo);
    ASYNC_B128(lbb + 16u, bbase, bo + 16u);
  };

  const int ntiles = K >> 5;
  stage(0, 0);   // prefetch tile 0

  for (int t = 0; t < ntiles; ++t) {
    WAIT_ASYNC0();     // my DMA for tile t has landed in LDS
    __syncthreads();   // everyone's tile t visible; buf[(t+1)&1] free of readers
    if (t + 1 < ntiles) stage((t + 1) & 1, (t + 1) << 5);  // overlap with compute

    const unsigned short* la_ = lds_a[t & 1];
    const unsigned short* lb_ = lds_b[t & 1];

    BFrag af[4];
#pragma unroll
    for (int mt = 0; mt < 4; mt++) {
      const __bf16* p = (const __bf16*)(la_ + (wm * 64 + mt * 16 + mrow) * 32);
      af[mt].h[0] = *(const v8bf*)(p + khalf * 16);
      af[mt].h[1] = *(const v8bf*)(p + khalf * 16 + 8);
    }
    v16bf bf[2];
#pragma unroll
    for (int nt = 0; nt < 2; nt++)
      bf[nt] = *(const v16bf*)(lb_ + (wn * 32 + nt * 16 + mrow) * 32 + khalf * 16);

#pragma unroll
    for (int mt = 0; mt < 4; mt++)
#pragma unroll
      for (int nt = 0; nt < 2; nt++)
        acc[mt][nt] = wmma_bf16(af[mt].v, bf[nt], acc[mt][nt]);
  }

  // epilogue: C element (row = v + 8*hi, col = lane&15) per tile
#pragma unroll
  for (int mt = 0; mt < 4; mt++) {
#pragma unroll
    for (int nt = 0; nt < 2; nt++) {
#pragma unroll
      for (int v = 0; v < 8; v++) {
        const int row = m0 + wm * 64 + mt * 16 + v + 8 * khalf;
        const int col = n0 + wn * 32 + nt * 16 + mrow;
        const float val = acc[mt][nt][v] + bias[col];
        if (mode == 0) {
          ((unsigned short*)Cout)[(size_t)row * N + col] = f2bf(val);
        } else if (mode == 1) {
          ((float*)Cout)[(size_t)row * N + col] = val;
        } else {
          const int bb = row / S, s = row % S;
          const int hh = col >> 6, dim = col & 63;
          ((unsigned short*)Cout)[(((size_t)bb * H + hh) * 64 + dim) * S + s] =
              f2bf(val);
        }
      }
    }
  }
}

// ---------------------------------------------------------------------------
// Flash attention (causal, online softmax).
// Grid: B*H*(S/64) blocks of 128 threads (4 waves). Each wave: 16 query rows.
// Q[b,s,h*64+d] bf16, K same layout, V transposed [b][h][d][s] bf16.
// ---------------------------------------------------------------------------
__global__ __launch_bounds__(128) void k_flash(
    const unsigned short* __restrict__ Qb, const unsigned short* __restrict__ Kb,
    const unsigned short* __restrict__ Vt, unsigned short* __restrict__ Ob,
    int Bn, int H, int S, int D, float scale) {
  __shared__ __attribute__((aligned(32))) unsigned short pbuf[4][16 * 32];

  const int wave = threadIdx.x >> 5, lane = threadIdx.x & 31;
  const int nq = S >> 6;
  const int qblk = blockIdx.x % nq;
  const int bh = blockIdx.x / nq;
  const int b = bh / H, h = bh % H;
  const int q0 = qblk * 64 + wave * 16;
  const int mrow = lane & 15, khalf = lane >> 4, kb16 = khalf * 16;

  // Q fragments (16 rows x 64 dims = 2 k-steps), resident for whole kernel
  BFrag aq[2];
  {
    const __bf16* qp =
        (const __bf16*)Qb + ((size_t)(b * S + q0 + mrow)) * D + h * 64;
#pragma unroll
    for (int ks = 0; ks < 2; ks++) {
      aq[ks].h[0] = *(const v8bf*)(qp + ks * 32 + khalf * 16);
      aq[ks].h[1] = *(const v8bf*)(qp + ks * 32 + khalf * 16 + 8);
    }
  }

  float m_i[8], l_i[8];
  v8f oacc[4];
#pragma unroll
  for (int v = 0; v < 8; v++) { m_i[v] = -1e30f; l_i[v] = 0.f; }
#pragma unroll
  for (int nt = 0; nt < 4; nt++) oacc[nt] = (v8f){0, 0, 0, 0, 0, 0, 0, 0};

  const int nblocks = (q0 + 47) >> 5;   // key blocks of 32, j0 <= q0 always
  for (int jb = 0; jb < nblocks; jb++) {
    const int j0 = jb * 32;

    // S = Q K^T for 32 keys: two 16-col C tiles, 2 k-steps each
    v8f s0a = (v8f){0, 0, 0, 0, 0, 0, 0, 0};
    v8f s1a = (v8f){0, 0, 0, 0, 0, 0, 0, 0};
#pragma unroll
    for (int ks = 0; ks < 2; ks++) {
      const __bf16* kp0 =
          (const __bf16*)Kb + ((size_t)(b * S + j0 + mrow)) * D + h * 64 + ks * 32 + kb16;
      s0a = wmma_bf16(aq[ks].v, *(const v16bf*)kp0, s0a);
      const __bf16* kp1 =
          (const __bf16*)Kb + ((size_t)(b * S + j0 + 16 + mrow)) * D + h * 64 + ks * 32 + kb16;
      s1a = wmma_bf16(aq[ks].v, *(const v16bf*)kp1, s1a);
    }

    // online softmax per row (row = v + 8*khalf, col = mrow within tile)
#pragma unroll
    for (int v = 0; v < 8; v++) {
      const int row = q0 + v + 8 * khalf;
      float s0 = s0a[v] * scale;
      float s1 = s1a[v] * scale;
      if (j0 + mrow > row)      s0 = -1e30f;   // causal mask
      if (j0 + 16 + mrow > row) s1 = -1e30f;
      const float mv = red16_max(fmaxf(s0, s1));
      const float mn = fmaxf(m_i[v], mv);
      const float alpha = __expf(m_i[v] - mn);
      const float p0 = __expf(s0 - mn);
      const float p1 = __expf(s1 - mn);
      const float rs = red16_sum(p0 + p1);
      l_i[v] = l_i[v] * alpha + rs;
      m_i[v] = mn;
#pragma unroll
      for (int nt = 0; nt < 4; nt++) oacc[nt][v] *= alpha;
      // relayout C->A via private LDS tile (16x32 bf16, row-major)
      const int prow = v + 8 * khalf;
      pbuf[wave][prow * 32 + mrow]      = f2bf(p0);
      pbuf[wave][prow * 32 + 16 + mrow] = f2bf(p1);
    }

    // P (16x32) A-fragment from LDS
    BFrag pf;
    const __bf16* pp = (const __bf16*)(&pbuf[wave][mrow * 32]);
    pf.h[0] = *(const v8bf*)(pp + khalf * 16);
    pf.h[1] = *(const v8bf*)(pp + khalf * 16 + 8);

    // O += P * V : V^T layout makes B-fragments contiguous along keys
#pragma unroll
    for (int nt = 0; nt < 4; nt++) {
      const __bf16* vp = (const __bf16*)Vt +
          ((size_t)((b * H + h) * 64 + nt * 16 + mrow)) * S + j0 + kb16;
      oacc[nt] = wmma_bf16(pf.v, *(const v16bf*)vp, oacc[nt]);
    }
  }

  // normalize and store bf16 to [b, s, h*64+d]
#pragma unroll
  for (int v = 0; v < 8; v++) {
    const float inv_l = 1.0f / l_i[v];
    const int row = q0 + v + 8 * khalf;
#pragma unroll
    for (int nt = 0; nt < 4; nt++) {
      const int col = h * 64 + nt * 16 + mrow;
      Ob[((size_t)(b * S + row)) * D + col] = f2bf(oacc[nt][v] * inv_l);
    }
  }
}

// ---------------------------------------------------------------------------
extern "C" void kernel_launch(void* const* d_in, const int* in_sizes, int n_in,
                              void* d_out, int out_size, void* d_ws, size_t ws_size,
                              hipStream_t stream) {
  (void)in_sizes; (void)n_in; (void)out_size; (void)ws_size;
  const float* x  = (const float*)d_in[0];
  const float* Wq = (const float*)d_in[1];
  const float* bq = (const float*)d_in[2];
  const float* Wk = (const float*)d_in[3];
  const float* bk = (const float*)d_in[4];
  const float* Wv = (const float*)d_in[5];
  const float* bv = (const float*)d_in[6];
  const float* Wo = (const float*)d_in[7];
  const float* bo = (const float*)d_in[8];
  float* out = (float*)d_out;

  const int B = 2, S = 2048, D = 1024, H = 16;
  const int M = B * S;                       // 4096
  const size_t MB = 1024 * 1024;
  char* ws = (char*)d_ws;
  unsigned short* xb  = (unsigned short*)(ws);            // 8 MB  x  bf16
  unsigned short* wqT = (unsigned short*)(ws +  8 * MB);  // 2 MB each
  unsigned short* wkT = (unsigned short*)(ws + 10 * MB);
  unsigned short* wvT = (unsigned short*)(ws + 12 * MB);
  unsigned short* woT = (unsigned short*)(ws + 14 * MB);
  unsigned short* Qb  = (unsigned short*)(ws + 16 * MB);  // 8 MB
  unsigned short* Kb  = (unsigned short*)(ws + 24 * MB);  // 8 MB
  unsigned short* Vt  = (unsigned short*)(ws + 32 * MB);  // 8 MB (per-head transposed)
  unsigned short* Ob  = (unsigned short*)(ws + 40 * MB);  // 8 MB

  k_cvt<<<(M * D + 255) / 256, 256, 0, stream>>>(x, xb, M * D);
  k_cvt_wt<<<(D * D + 255) / 256, 256, 0, stream>>>(Wq, wqT, D, D);
  k_cvt_wt<<<(D * D + 255) / 256, 256, 0, stream>>>(Wk, wkT, D, D);
  k_cvt_wt<<<(D * D + 255) / 256, 256, 0, stream>>>(Wv, wvT, D, D);
  k_cvt_wt<<<(D * D + 255) / 256, 256, 0, stream>>>(Wo, woT, D, D);

  dim3 gg(M / 128, D / 128);
  k_gemm_bf16<<<gg, 256, 0, stream>>>(xb, wqT, bq, Qb, M, D, D, 0, H, S);
  k_gemm_bf16<<<gg, 256, 0, stream>>>(xb, wkT, bk, Kb, M, D, D, 0, H, S);
  k_gemm_bf16<<<gg, 256, 0, stream>>>(xb, wvT, bv, Vt, M, D, D, 2, H, S);

  k_flash<<<B * H * (S / 64), 128, 0, stream>>>(Qb, Kb, Vt, Ob, B, H, S, D,
                                                0.03125f /* DIM^-0.5 */);

  k_gemm_bf16<<<gg, 256, 0, stream>>>(Ob, woT, bo, out, M, D, D, 1, H, S);
}
